// Wasserstein_22986664968862
// MI455X (gfx1250) — matrix operand order repaired and checked by reading, compiled, not run on hardware
//
#include <hip/hip_runtime.h>

// ---------------------------------------------------------------------------
// Sinkhorn divergence (geomloss, cosine^2 cost) for MI455X / gfx1250.
// Fused flash-style softmin: bf16-split WMMA GEMM + online logsumexp.
// ---------------------------------------------------------------------------

typedef __attribute__((ext_vector_type(16))) __bf16       bf16x16;
typedef __attribute__((ext_vector_type(8)))  float        f32x8;
typedef __attribute__((ext_vector_type(4)))  unsigned int u32x4;
typedef __attribute__((ext_vector_type(2)))  float        f32x2;

#define NPTS      4096
#define DIM       64
#define NBATCH    2
#define MAX_EPS   32
#define MAX_ITERS 30

struct Ctrl {
    int   n_eps;
    int   pad;
    float eps[MAX_EPS];
};

union Frag {
    u32x4   q[2];
    bf16x16 v;
};

__device__ __forceinline__ unsigned short f2bf(float f) {
    unsigned u = __float_as_uint(f);
    u += 0x7FFFu + ((u >> 16) & 1u);          // RNE round to bf16
    return (unsigned short)(u >> 16);
}
__device__ __forceinline__ float bf2f(unsigned short h) {
    return __uint_as_float(((unsigned)h) << 16);
}

// --------------------------------------------------------------------------
// 1) Normalize each point (f32) and split into bf16 hi/lo pairs.
//    One wave (32 lanes) per 64-element row; 4 rows per 128-thread block.
// --------------------------------------------------------------------------
__global__ __launch_bounds__(128) void normalize_split_kernel(
    const float* __restrict__ x, const float* __restrict__ y,
    unsigned short* __restrict__ xhi, unsigned short* __restrict__ xlo,
    unsigned short* __restrict__ yhi, unsigned short* __restrict__ ylo) {
    const int wave = threadIdx.x >> 5, lane = threadIdx.x & 31;
    const int row  = blockIdx.x * 4 + wave;
    if (row >= 2 * NBATCH * NPTS) return;

    const float* src; unsigned short* dhi; unsigned short* dlo;
    if (row < NBATCH * NPTS) {
        src = x + (size_t)row * DIM;
        dhi = xhi + (size_t)row * DIM; dlo = xlo + (size_t)row * DIM;
    } else {
        const int r = row - NBATCH * NPTS;
        src = y + (size_t)r * DIM;
        dhi = yhi + (size_t)r * DIM; dlo = ylo + (size_t)r * DIM;
    }
    f32x2 v = *(const f32x2*)(src + 2 * lane);
    float s = v.x * v.x + v.y * v.y;
    #pragma unroll
    for (int m = 16; m >= 1; m >>= 1) s += __shfl_xor(s, m, 32);
    const float rn = 1.0f / sqrtf(s);
    const float a0 = v.x * rn, a1 = v.y * rn;
    const unsigned short h0 = f2bf(a0), h1 = f2bf(a1);
    dhi[2 * lane]     = h0;
    dhi[2 * lane + 1] = h1;
    dlo[2 * lane]     = f2bf(a0 - bf2f(h0));
    dlo[2 * lane + 1] = f2bf(a1 - bf2f(h1));
}

// --------------------------------------------------------------------------
// 2a) Per-block min/max partials over the concatenated (raw) clouds.
//     64 blocks x 256 points each; thread = (point-group, coordinate).
// --------------------------------------------------------------------------
__global__ __launch_bounds__(256) void minmax_part_kernel(
    const float* __restrict__ x, const float* __restrict__ y,
    float* __restrict__ pmin, float* __restrict__ pmax) {
    const int c   = threadIdx.x & 63;
    const int grp = threadIdx.x >> 6;              // 0..3
    float mn = 3.402823466e38f, mx = -3.402823466e38f;
    const int base = blockIdx.x * 256;
    for (int i = grp; i < 256; i += 4) {
        const int p = base + i;
        const float* src = (p < NBATCH * NPTS) ? (x + (size_t)p * DIM)
                                               : (y + (size_t)(p - NBATCH * NPTS) * DIM);
        const float v = src[c];
        mn = fminf(mn, v); mx = fmaxf(mx, v);
    }
    __shared__ float smn[256], smx[256];
    smn[threadIdx.x] = mn; smx[threadIdx.x] = mx;
    __syncthreads();
    if (grp == 0) {
        #pragma unroll
        for (int g2 = 1; g2 < 4; ++g2) {
            mn = fminf(mn, smn[g2 * 64 + c]);
            mx = fmaxf(mx, smx[g2 * 64 + c]);
        }
        pmin[blockIdx.x * 64 + c] = mn;
        pmax[blockIdx.x * 64 + c] = mx;
    }
}

// --------------------------------------------------------------------------
// 2b) Reduce partials -> diameter -> eps schedule (geomloss epsilon_schedule)
// --------------------------------------------------------------------------
__global__ __launch_bounds__(64) void schedule_kernel(
    const float* __restrict__ pmin, const float* __restrict__ pmax,
    Ctrl* __restrict__ ctrl) {
    const int c = threadIdx.x;
    float mn = 3.402823466e38f, mx = -3.402823466e38f;
    for (int b2 = 0; b2 < 64; ++b2) {
        mn = fminf(mn, pmin[b2 * 64 + c]);
        mx = fmaxf(mx, pmax[b2 * 64 + c]);
    }
    __shared__ float s[64];
    const float d = mx - mn;
    s[c] = d * d;
    __syncthreads();
    if (c == 0) {
        float d2 = 0.0f;
        for (int i = 0; i < 64; ++i) d2 += s[i];   // diameter^2 = eps_list[0]
        int n = 0;
        ctrl->eps[n++] = d2;
        float ln = __logf(d2);                      // 2*log(diameter)
        const float step = -1.3862943611198906f;    // 2*log(0.5)
        const float stop = -2.3025850929940455f;    // 2*log(blur) = log(0.1)
        while (ln > stop && n < MAX_EPS - 2) {      // np.arange(2logd, 2logblur, 2log.5)
            ctrl->eps[n++] = __expf(ln);
            ln += step;
        }
        ctrl->eps[n++] = 0.1f;                      // blur^2
        ctrl->n_eps = n;
    }
}

// --------------------------------------------------------------------------
// 3) Fused softmin pass:  out_i = -eps * logsumexp_j( h_j - (1 - <xi,yj>)^2/eps )
//    h_j = -log(4096) + (useG ? g_j/eps : 0)
//    Each wave owns 16 rows; A fragments (bf16 hi/lo, K=64) are resident;
//    loop over 256 column tiles with 6 chained v_wmma_f32_16x16x32_bf16
//    (hi*hi + hi*lo + lo*hi split) and a 1-exp online logsumexp epilogue.
//    iterMode: -2 = init (eps[0]),  -1 = final (eps[n-1]),  k>=0 = guarded.
// --------------------------------------------------------------------------
__global__ __launch_bounds__(128) void softmin_wmma_kernel(
    const unsigned short* __restrict__ rowHi, const unsigned short* __restrict__ rowLo,
    const unsigned short* __restrict__ colHi, const unsigned short* __restrict__ colLo,
    const float* __restrict__ g, float* __restrict__ outp,
    const Ctrl* __restrict__ ctrl, int iterMode, int useG) {
    const int neps = ctrl->n_eps;
    if (iterMode >= 0 && iterMode >= neps) return;
    const int ei   = (iterMode == -2) ? 0 : ((iterMode == -1) ? (neps - 1) : iterMode);
    const float eps     = ctrl->eps[ei];
    const float inv_eps = 1.0f / eps;

    const int b = blockIdx.z;
    const size_t nd = (size_t)NPTS * DIM;
    rowHi += b * nd; rowLo += b * nd; colHi += b * nd; colLo += b * nd;
    g    += (size_t)b * NPTS;
    outp += (size_t)b * NPTS;

    const int wave = threadIdx.x >> 5, lane = threadIdx.x & 31;
    const int row0 = blockIdx.x * 64 + wave * 16;
    const int m    = lane & 15, half = lane >> 4;

    // A fragments: 16x32 bf16 layout — lane(m,half): K runs [8h..8h+7], [16+8h..16+8h+7]
    Frag aH0, aH1, aL0, aL1;
    {
        const unsigned short* rb = rowHi + (size_t)(row0 + m) * DIM;
        const unsigned short* rl = rowLo + (size_t)(row0 + m) * DIM;
        aH0.q[0] = *(const u32x4*)(rb + 8 * half);
        aH0.q[1] = *(const u32x4*)(rb + 16 + 8 * half);
        aH1.q[0] = *(const u32x4*)(rb + 32 + 8 * half);
        aH1.q[1] = *(const u32x4*)(rb + 48 + 8 * half);
        aL0.q[0] = *(const u32x4*)(rl + 8 * half);
        aL0.q[1] = *(const u32x4*)(rl + 16 + 8 * half);
        aL1.q[0] = *(const u32x4*)(rl + 32 + 8 * half);
        aL1.q[1] = *(const u32x4*)(rl + 48 + 8 * half);
    }

    float mx[8], sm[8];
    #pragma unroll
    for (int v = 0; v < 8; ++v) { mx[v] = -3.402823466e38f; sm[v] = 0.0f; }
    const float hbase = -8.317766166719343f;   // -log(4096)

    for (int jt = 0; jt < NPTS / 16; ++jt) {
        const int col0 = jt << 4;
        // B fragments: 32x16 bf16 — lane(n,half) holds yj[n], K runs [8h..8h+7], [16+8h..16+8h+7]
        const unsigned short* cH = colHi + (size_t)(col0 + m) * DIM + 8 * half;
        const unsigned short* cL = colLo + (size_t)(col0 + m) * DIM + 8 * half;
        Frag bH0, bH1, bL0, bL1;
        bH0.q[0] = *(const u32x4*)(cH);       bH0.q[1] = *(const u32x4*)(cH + 16);
        bH1.q[0] = *(const u32x4*)(cH + 32);  bH1.q[1] = *(const u32x4*)(cH + 48);
        bL0.q[0] = *(const u32x4*)(cL);       bL0.q[1] = *(const u32x4*)(cL + 16);
        bL1.q[0] = *(const u32x4*)(cL + 32);  bL1.q[1] = *(const u32x4*)(cL + 48);
        if (jt + 1 < NPTS / 16) {
            __builtin_prefetch(cH + 16 * DIM, 0, 1);   // global_prefetch next tile
            __builtin_prefetch(cL + 16 * DIM, 0, 1);
        }

        f32x8 acc = {0.f, 0.f, 0.f, 0.f, 0.f, 0.f, 0.f, 0.f};
        acc = __builtin_amdgcn_wmma_f32_16x16x32_bf16(false, aH0.v, false, bH0.v, (short)0, acc, false, false);
        acc = __builtin_amdgcn_wmma_f32_16x16x32_bf16(false, aH1.v, false, bH1.v, (short)0, acc, false, false);
        acc = __builtin_amdgcn_wmma_f32_16x16x32_bf16(false, aL0.v, false, bH0.v, (short)0, acc, false, false);
        acc = __builtin_amdgcn_wmma_f32_16x16x32_bf16(false, aL1.v, false, bH1.v, (short)0, acc, false, false);
        acc = __builtin_amdgcn_wmma_f32_16x16x32_bf16(false, aH0.v, false, bL0.v, (short)0, acc, false, false);
        acc = __builtin_amdgcn_wmma_f32_16x16x32_bf16(false, aH1.v, false, bL1.v, (short)0, acc, false, false);

        float hval = hbase;
        if (useG) hval = fmaf(g[col0 + m], inv_eps, hbase);

        #pragma unroll
        for (int v = 0; v < 8; ++v) {
            const float u = 1.0f - acc[v];                 // 1 - cos
            const float t = fmaf(-u * u, inv_eps, hval);   // h - C/eps
            const float d = t - mx[v];
            const float e = __expf(-fabsf(d));             // single exp / element
            const bool  gt = d > 0.0f;
            sm[v] = gt ? fmaf(sm[v], e, 1.0f) : (sm[v] + e);
            mx[v] = gt ? t : mx[v];
        }
    }

    // Row results live across the 16 lanes of each half: combine (max, sum)
    #pragma unroll
    for (int v = 0; v < 8; ++v) {
        #pragma unroll
        for (int off = 1; off < 16; off <<= 1) {
            const float om = __shfl_xor(mx[v], off, 32);
            const float os = __shfl_xor(sm[v], off, 32);
            const float nm = fmaxf(mx[v], om);
            sm[v] = sm[v] * __expf(mx[v] - nm) + os * __expf(om - nm);
            mx[v] = nm;
        }
    }
    if (m == 0) {      // lanes 0 and 16: rows row0 + 8*half + v
        #pragma unroll
        for (int v = 0; v < 8; ++v)
            outp[row0 + 8 * half + v] = -eps * (mx[v] + __logf(sm[v]));
    }
}

// --------------------------------------------------------------------------
// 4) Symmetrized averaging update  f <- 0.5*(f + ft)  (guarded by schedule)
// --------------------------------------------------------------------------
__global__ __launch_bounds__(256) void avg_kernel(
    float* __restrict__ f0, const float* __restrict__ t0,
    float* __restrict__ f1, const float* __restrict__ t1,
    float* __restrict__ f2, const float* __restrict__ t2,
    float* __restrict__ f3, const float* __restrict__ t3,
    const Ctrl* __restrict__ ctrl, int iter) {
    if (iter >= ctrl->n_eps) return;
    const int i = blockIdx.x * 256 + threadIdx.x;
    if (i >= NBATCH * NPTS) return;
    f0[i] = 0.5f * (f0[i] + t0[i]);
    f1[i] = 0.5f * (f1[i] + t1[i]);
    f2[i] = 0.5f * (f2[i] + t2[i]);
    f3[i] = 0.5f * (f3[i] + t3[i]);
}

// --------------------------------------------------------------------------
// 5) Debiased sinkhorn cost:  out_b = mean_i(fba-faa) + mean_j(gab-gbb)
// --------------------------------------------------------------------------
__global__ __launch_bounds__(256) void cost_kernel(
    const float* __restrict__ fba, const float* __restrict__ faa,
    const float* __restrict__ gab, const float* __restrict__ gbb,
    float* __restrict__ out) {
    const int b = blockIdx.x;
    __shared__ float red[256];
    float s = 0.0f;
    const size_t o = (size_t)b * NPTS;
    for (int i = threadIdx.x; i < NPTS; i += 256)
        s += (fba[o + i] - faa[o + i]) + (gab[o + i] - gbb[o + i]);
    red[threadIdx.x] = s;
    __syncthreads();
    for (int st = 128; st > 0; st >>= 1) {
        if (threadIdx.x < st) red[threadIdx.x] += red[threadIdx.x + st];
        __syncthreads();
    }
    if (threadIdx.x == 0) out[b] = red[0] * (1.0f / NPTS);
}

// --------------------------------------------------------------------------
// Host orchestration (all launches on `stream`, graph-capture safe)
// --------------------------------------------------------------------------
extern "C" void kernel_launch(void* const* d_in, const int* in_sizes, int n_in,
                              void* d_out, int out_size, void* d_ws, size_t ws_size,
                              hipStream_t stream) {
    (void)in_sizes; (void)n_in; (void)out_size; (void)ws_size;
    const float* x = (const float*)d_in[0];   // (B,N,D) f32
    const float* y = (const float*)d_in[1];   // (B,N,D) f32
    float* out = (float*)d_out;               // (B,) f32

    char* w = (char*)d_ws;
    size_t off = 0;
    auto alloc = [&](size_t bytes) -> char* {
        char* p = w + off;
        off += (bytes + 255) & ~(size_t)255;
        return p;
    };
    const size_t NDB = (size_t)NBATCH * NPTS * DIM * sizeof(unsigned short);
    const size_t NB  = (size_t)NBATCH * NPTS * sizeof(float);

    Ctrl*  ctrl = (Ctrl*)alloc(sizeof(Ctrl));
    float* pmin = (float*)alloc(64 * 64 * sizeof(float));
    float* pmax = (float*)alloc(64 * 64 * sizeof(float));
    unsigned short* xhi = (unsigned short*)alloc(NDB);
    unsigned short* xlo = (unsigned short*)alloc(NDB);
    unsigned short* yhi = (unsigned short*)alloc(NDB);
    unsigned short* ylo = (unsigned short*)alloc(NDB);
    float* f_ba  = (float*)alloc(NB);
    float* g_ab  = (float*)alloc(NB);
    float* f_aa  = (float*)alloc(NB);
    float* g_bb  = (float*)alloc(NB);
    float* ft_ba = (float*)alloc(NB);
    float* gt_ab = (float*)alloc(NB);
    float* ft_aa = (float*)alloc(NB);
    float* gt_bb = (float*)alloc(NB);

    // normalize + bf16 split; diameter + eps schedule (device side)
    normalize_split_kernel<<<(2 * NBATCH * NPTS) / 4, 128, 0, stream>>>(x, y, xhi, xlo, yhi, ylo);
    minmax_part_kernel<<<64, 256, 0, stream>>>(x, y, pmin, pmax);
    schedule_kernel<<<1, 64, 0, stream>>>(pmin, pmax, ctrl);

    const dim3 sg(NPTS / 64, 1, NBATCH);

    // init at eps_list[0]  (h = log-uniform weights only)
    softmin_wmma_kernel<<<sg, 128, 0, stream>>>(xhi, xlo, xhi, xlo, f_aa, f_aa, ctrl, -2, 0);
    softmin_wmma_kernel<<<sg, 128, 0, stream>>>(yhi, ylo, yhi, ylo, g_bb, g_bb, ctrl, -2, 0);
    softmin_wmma_kernel<<<sg, 128, 0, stream>>>(yhi, ylo, xhi, xlo, g_ab, g_ab, ctrl, -2, 0);
    softmin_wmma_kernel<<<sg, 128, 0, stream>>>(xhi, xlo, yhi, ylo, f_ba, f_ba, ctrl, -2, 0);

    // eps-scaling loop (fixed launch count, device-guarded by schedule length)
    for (int it = 0; it < MAX_ITERS; ++it) {
        softmin_wmma_kernel<<<sg, 128, 0, stream>>>(xhi, xlo, yhi, ylo, g_ab, ft_ba, ctrl, it, 1);
        softmin_wmma_kernel<<<sg, 128, 0, stream>>>(yhi, ylo, xhi, xlo, f_ba, gt_ab, ctrl, it, 1);
        softmin_wmma_kernel<<<sg, 128, 0, stream>>>(xhi, xlo, xhi, xlo, f_aa, ft_aa, ctrl, it, 1);
        softmin_wmma_kernel<<<sg, 128, 0, stream>>>(yhi, ylo, yhi, ylo, g_bb, gt_bb, ctrl, it, 1);
        avg_kernel<<<(NBATCH * NPTS + 255) / 256, 256, 0, stream>>>(
            f_ba, ft_ba, g_ab, gt_ab, f_aa, ft_aa, g_bb, gt_bb, ctrl, it);
    }

    // final extrapolation at eps_list[-1] (reads pre-update duals)
    softmin_wmma_kernel<<<sg, 128, 0, stream>>>(xhi, xlo, yhi, ylo, g_ab, ft_ba, ctrl, -1, 1);
    softmin_wmma_kernel<<<sg, 128, 0, stream>>>(yhi, ylo, xhi, xlo, f_ba, gt_ab, ctrl, -1, 1);
    softmin_wmma_kernel<<<sg, 128, 0, stream>>>(xhi, xlo, xhi, xlo, f_aa, ft_aa, ctrl, -1, 1);
    softmin_wmma_kernel<<<sg, 128, 0, stream>>>(yhi, ylo, yhi, ylo, g_bb, gt_bb, ctrl, -1, 1);

    cost_kernel<<<NBATCH, 256, 0, stream>>>(ft_ba, ft_aa, gt_ab, gt_bb, out);
}
